// GraphRegularizer_77352361001298
// MI455X (gfx1250) — matrix operand order repaired and checked by reading, not compiled, not run on hardware
//
#include <hip/hip_runtime.h>
#include <hip/hip_bf16.h>

typedef _Float16 v16h __attribute__((ext_vector_type(16)));
typedef _Float16 v8h  __attribute__((ext_vector_type(8)));
typedef float    v8f  __attribute__((ext_vector_type(8)));
typedef float    f4v  __attribute__((ext_vector_type(4)));

#define DREF 128           // node embedding dim (reference fixed)
#define KDIM 256           // 2*DREF
#define LDS_W1_ELEMS 32768 // 128*256 f16, pre-swizzled into B-fragment layout (64KB)

// ---------------------------------------------------------------------------
// Kernel 1: zero-fill adjacency (1 GiB) with nontemporal 16B stores.
// ---------------------------------------------------------------------------
__global__ __launch_bounds__(256) void gr_zero_kernel(f4v* __restrict__ out, long long n4) {
    long long i = (long long)blockIdx.x * blockDim.x + threadIdx.x;
    long long stride = (long long)gridDim.x * blockDim.x;
    f4v z = {0.0f, 0.0f, 0.0f, 0.0f};
    for (; i < n4; i += stride) {
        __builtin_nontemporal_store(z, &out[i]);
    }
}

// ---------------------------------------------------------------------------
// Kernel 2: edge MLP via v_wmma_f32_16x16x32_f16.
//   One wave per 16-edge tile. X = [16 edges x 256] (concat of two gathered
//   embeddings), W1 = [128 out x 256 in]. H = X*W1^T  ->  8 K-steps x 8 N-tiles.
//   W1 staged in LDS as f16, pre-swizzled into per-(kstep,ntile) B fragments:
//   fragment (kb,t): lane L holds B(k = 16*(j>>3) + 8*(L>>4) + (j&7), n = L&15), j=0..15,
//   i.e. W1[16t + (L&15)][32kb + 16*(j>>3) + 8*(L>>4) + (j&7)]  -> 16 contiguous f16 per lane.
//   The fragment offset is laundered through an empty asm so the compiler
//   cannot hoist all 64 fragments out of the tile loop (which previously
//   caused a 512-VGPR scratch spill feeding the WMMAs from scratch memory).
// ---------------------------------------------------------------------------
__global__ __launch_bounds__(256) void gr_edge_mlp_kernel(
    const float* __restrict__ emb,        // [N,128] fp32
    const long long* __restrict__ eidx,   // [2,E] int64
    const float* __restrict__ W1,         // [128,256] fp32
    const float* __restrict__ b1,         // [128]
    const float* __restrict__ W2,         // [1,128]
    const float* __restrict__ b2,         // [1]
    float* __restrict__ adj,              // [N,N]
    int N, long long E)
{
    __shared__ _Float16 sW1[LDS_W1_ELEMS];   // exactly 64KB

    // Convert + swizzle W1 into B-fragment layout (coalesced global reads).
    for (int i = threadIdx.x; i < LDS_W1_ELEMS; i += 256) {
        int j    = i & 15;
        int lane = (i >> 4) & 31;
        int t    = (i >> 9) & 7;
        int kb   = i >> 12;
        int o    = 16 * t + (lane & 15);                 // output feature (B column)
        int in   = 32 * kb + ((j >> 3) << 4) + ((lane >> 4) << 3) + (j & 7);     // K index
        sW1[i] = (_Float16)W1[o * KDIM + in];
    }
    __syncthreads();

    const int lane = threadIdx.x & 31;
    const int wave = threadIdx.x >> 5;
    const int hi   = lane >> 4;        // lane half
    const int mlo  = lane & 15;
    const float bias2 = b2[0];
    const long long Nll = (long long)N;

    const long long nTiles = E >> 4;
    for (long long tile = (long long)blockIdx.x * 8 + wave; tile < nTiles;
         tile += (long long)gridDim.x * 8) {

        // This lane gathers row m = mlo of the X tile.
        const long long e0 = tile * 16 + mlo;
        const long long r  = eidx[e0];
        const long long c  = eidx[E + e0];
        const float* rowp = emb + r * DREF;
        const float* colp = emb + c * DREF;

        v8f acc[8] = {};   // 8 N-tiles x (16x16 f32 D-fragment)

        #pragma unroll
        for (int kb = 0; kb < 8; ++kb) {
            // A fragment: lanes 0-15 hold K = kb*32 + {0..7, 16..23},
            //             lanes 16-31 hold K = kb*32 + {8..15, 24..31}.
            const int k0 = kb * 32 + hi * 8;        // first 8-element segment
            const int k1 = kb * 32 + 16 + hi * 8;   // second 8-element segment
            const float* s0 = (k0 < DREF) ? (rowp + k0) : (colp + (k0 - DREF));
            const float* s1 = (k1 < DREF) ? (rowp + k1) : (colp + (k1 - DREF));
            float4 f0a = *(const float4*)(s0);
            float4 f0b = *(const float4*)(s0 + 4);
            float4 f1a = *(const float4*)(s1);
            float4 f1b = *(const float4*)(s1 + 4);

            union { v16h v; _Float16 e[16]; } A;
            A.e[0]  = (_Float16)f0a.x; A.e[1]  = (_Float16)f0a.y;
            A.e[2]  = (_Float16)f0a.z; A.e[3]  = (_Float16)f0a.w;
            A.e[4]  = (_Float16)f0b.x; A.e[5]  = (_Float16)f0b.y;
            A.e[6]  = (_Float16)f0b.z; A.e[7]  = (_Float16)f0b.w;
            A.e[8]  = (_Float16)f1a.x; A.e[9]  = (_Float16)f1a.y;
            A.e[10] = (_Float16)f1a.z; A.e[11] = (_Float16)f1a.w;
            A.e[12] = (_Float16)f1b.x; A.e[13] = (_Float16)f1b.y;
            A.e[14] = (_Float16)f1b.z; A.e[15] = (_Float16)f1b.w;

            #pragma unroll
            for (int t = 0; t < 8; ++t) {
                int off = (((kb * 8 + t) * 32 + lane) << 4);
                // Launder the LDS offset: prevents LICM/CSE of the fragment
                // loads across iterations (avoids the 512-VGPR scratch spill).
                asm volatile("" : "+v"(off));
                const _Float16* bp = sW1 + off;
                union { v16h v; v8h h[2]; } B;
                B.h[0] = *(const v8h*)(bp);
                B.h[1] = *(const v8h*)(bp + 8);
                acc[t] = __builtin_amdgcn_wmma_f32_16x16x32_f16(
                    /*neg_a=*/false, A.v, /*neg_b=*/false, B.v,
                    /*c_mod=*/(short)0, acc[t],
                    /*reuse_a=*/false, /*reuse_b=*/false);
            }
        }

        // Epilogue: h = ELU(acc + b1); partial = h * W2; reduce over features.
        // D fragment: acc[t][v] in lane L = H(edge m = v + 8*hi, feat n = 16t + mlo).
        float s[8] = {0.f, 0.f, 0.f, 0.f, 0.f, 0.f, 0.f, 0.f};
        #pragma unroll
        for (int t = 0; t < 8; ++t) {
            const float b1v = b1[16 * t + mlo];
            const float w2v = W2[16 * t + mlo];
            #pragma unroll
            for (int v = 0; v < 8; ++v) {
                float hv = acc[t][v] + b1v;
                float ev = (hv > 0.0f) ? hv : (__expf(hv) - 1.0f);
                s[v] += ev * w2v;
            }
        }

        #pragma unroll
        for (int v = 0; v < 8; ++v) {
            float x = s[v];
            x += __shfl_xor(x, 1, 32);
            x += __shfl_xor(x, 2, 32);
            x += __shfl_xor(x, 4, 32);
            x += __shfl_xor(x, 8, 32);   // full sum within each 16-lane half
            if (mlo == v) {              // one storer per edge per half
                const long long eS = tile * 16 + v + 8 * hi;
                const long long r2 = eidx[eS];
                const long long c2 = eidx[E + eS];
                const float p = 1.0f / (1.0f + __expf(-(x + bias2)));
                adj[r2 * Nll + c2] = p;
            }
        }
    }
}

// ---------------------------------------------------------------------------
// Kernel 2b: scalar tail for E % 16 != 0 (not hit for E = 524288).
// ---------------------------------------------------------------------------
__global__ __launch_bounds__(128) void gr_edge_tail_kernel(
    const float* __restrict__ emb, const long long* __restrict__ eidx,
    const float* __restrict__ W1, const float* __restrict__ b1,
    const float* __restrict__ W2, const float* __restrict__ b2,
    float* __restrict__ adj, int N, long long E, long long start)
{
    const long long e = start + blockIdx.x;
    if (e >= E) return;
    __shared__ float sh[DREF];
    const long long r = eidx[e], c = eidx[E + e];
    const int f = threadIdx.x;                // one output feature per thread
    float acc = b1[f];
    const float* wrow = W1 + (long long)f * KDIM;
    for (int k = 0; k < DREF; ++k) acc += emb[r * DREF + k] * wrow[k];
    for (int k = 0; k < DREF; ++k) acc += emb[c * DREF + k] * wrow[DREF + k];
    float ev = (acc > 0.0f) ? acc : (__expf(acc) - 1.0f);
    sh[f] = ev * W2[f];
    __syncthreads();
    for (int sN = 64; sN > 0; sN >>= 1) {
        if (f < sN) sh[f] += sh[f + sN];
        __syncthreads();
    }
    if (f == 0) adj[r * (long long)N + c] = 1.0f / (1.0f + __expf(-(sh[0] + b2[0])));
}

// ---------------------------------------------------------------------------
// Kernel 3: diagonal = 1.0 (must run after the edge scatter).
// ---------------------------------------------------------------------------
__global__ __launch_bounds__(256) void gr_diag_kernel(float* __restrict__ adj, int N) {
    int i = blockIdx.x * blockDim.x + threadIdx.x;
    if (i < N) adj[(long long)i * N + i] = 1.0f;
}

extern "C" void kernel_launch(void* const* d_in, const int* in_sizes, int n_in,
                              void* d_out, int out_size, void* d_ws, size_t ws_size,
                              hipStream_t stream) {
    const float*     emb  = (const float*)d_in[0];
    const long long* eidx = (const long long*)d_in[1];
    // d_in[2] = batch (unused by reference)
    const float* W1 = (const float*)d_in[3];
    const float* b1 = (const float*)d_in[4];
    const float* W2 = (const float*)d_in[5];
    const float* b2 = (const float*)d_in[6];
    float* adj = (float*)d_out;

    const long long N = in_sizes[2];          // batch has N elements
    const long long E = (long long)in_sizes[1] / 2;

    // 1) zero-fill adjacency (out_size = N*N, divisible by 4)
    const long long n4 = (long long)out_size / 4;
    gr_zero_kernel<<<4096, 256, 0, stream>>>((f4v*)adj, n4);

    // 2) edge MLP + scatter (WMMA path over full 16-edge tiles)
    const long long nTiles = E >> 4;
    if (nTiles > 0) {
        gr_edge_mlp_kernel<<<1024, 256, 0, stream>>>(emb, eidx, W1, b1, W2, b2,
                                                     adj, (int)N, E);
    }
    const long long tail = E - (nTiles << 4);
    if (tail > 0) {
        gr_edge_tail_kernel<<<(int)tail, 128, 0, stream>>>(emb, eidx, W1, b1, W2, b2,
                                                           adj, (int)N, E, nTiles << 4);
    }

    // 3) diagonal overrides any scattered edge probs (reference ordering)
    gr_diag_kernel<<<(int)((N + 255) / 256), 256, 0, stream>>>(adj, (int)N);
}
